// Fused_Gromov_Loss_57801669869628
// MI455X (gfx1250) — compile-verified, matrix operand order checked by reading, not run on hardware
//
#include <hip/hip_runtime.h>

// ---------------------------------------------------------------------------
// Fused Gromov loss for MI455X (gfx1250, wave32, WMMA).
// N=4096, D=512. Dominant cost: 10x 4096^3 GEMMs -> bf16 WMMA, f32 accum.
// Double-buffered LDS pipeline; async global->LDS copies (ASYNCcnt) when
// the toolchain exposes the builtin (it does here, per round-2 diagnostic).
// ---------------------------------------------------------------------------

#define NN 4096
#define DD 512
#define TILE 128
#define KSTEP 32
#define LDSROW 40 // ushort elements per LDS row (80B) -> conflict-free, 16B aligned

typedef __attribute__((ext_vector_type(16))) __bf16 v16bf;
typedef __attribute__((ext_vector_type(8)))  __bf16 v8bf;
typedef __attribute__((ext_vector_type(8)))  float  v8f;
typedef __attribute__((ext_vector_type(4)))  unsigned int u32x4;
typedef int v4i_vs __attribute__((vector_size(16))); // matches builtin param type

#if defined(__HIP_DEVICE_COMPILE__) && __has_builtin(__builtin_amdgcn_global_load_async_to_lds_b128)
#define ASYNC_LDS 1
#else
#define ASYNC_LDS 0
#endif

static __device__ __forceinline__ void g2l_b128(unsigned short* lds,
                                                const unsigned short* g) {
#if ASYNC_LDS
  __builtin_amdgcn_global_load_async_to_lds_b128((v4i_vs*)g, (v4i_vs*)lds, 0, 0);
#else
  (void)lds; (void)g;
#endif
}

static __device__ __forceinline__ void async_wait0() {
#if ASYNC_LDS
#if __has_builtin(__builtin_amdgcn_s_wait_asynccnt)
  __builtin_amdgcn_s_wait_asynccnt(0);
#else
  asm volatile("s_wait_asynccnt 0" ::: "memory");
#endif
#endif
}

static __device__ __forceinline__ unsigned short f32_bf16(float f) {
  union { float f; unsigned u; } v; v.f = f;
  unsigned u = v.u;
  u += 0x7FFFu + ((u >> 16) & 1u); // round-to-nearest-even
  return (unsigned short)(u >> 16);
}

// ---------------------------------------------------------------------------
__global__ void cvt_bf16_kernel(const float* __restrict__ src,
                                unsigned short* __restrict__ dst, int n) {
  int i = blockIdx.x * blockDim.x + threadIdx.x;
  int stride = gridDim.x * blockDim.x;
  for (; i < n; i += stride) dst[i] = f32_bf16(src[i]);
}

__global__ void zero_kernel(float* __restrict__ p, int n) {
  int i = blockIdx.x * blockDim.x + threadIdx.x;
  int stride = gridDim.x * blockDim.x;
  for (; i < n; i += stride) p[i] = 0.0f;
}

// ---------------------------------------------------------------------------
// Fragment: ISA 7.12.2 16-bit A/B layout. lane -> row/col = lane&15,
// kb = (lane>>4)*8, e[0..7] = K kb..kb+7, e[8..15] = K kb+16..kb+23.
// ---------------------------------------------------------------------------
static __device__ __forceinline__ v16bf load_frag(const unsigned short* p, int kb) {
  v8bf lo = *(const v8bf*)(p + kb);
  v8bf hi = *(const v8bf*)(p + kb + 16);
  v16bf r;
#pragma unroll
  for (int i = 0; i < 8; ++i) { r[i] = lo[i]; r[i + 8] = hi[i]; }
  return r;
}

// ---------------------------------------------------------------------------
// GEMM variant 1: D = A * B^T, K = 512 (feature gram matrices).
// mode 0: outB(bf16) = acc ; mode 1: outF(f32) = 1 - acc.
// Double-buffered LDS; 8 waves (2x4), 8 WMMA tiles per wave.
// ---------------------------------------------------------------------------
__global__ __launch_bounds__(256)
void gemm_abt_k512(const unsigned short* __restrict__ A,
                   const unsigned short* __restrict__ B,
                   unsigned short* __restrict__ outB,
                   float* __restrict__ outF, int mode) {
  __shared__ unsigned short As[2][TILE * LDSROW];
  __shared__ unsigned short Bs[2][TILE * LDSROW];
  const int t = threadIdx.x, lane = t & 31, wid = t >> 5;
  const int wm = wid >> 2, wn = wid & 3;
  const int rowBase = blockIdx.y * TILE, colBase = blockIdx.x * TILE;

  int rr[2], kk[2];
#pragma unroll
  for (int j = 0; j < 2; ++j) { int c = t + 256 * j; rr[j] = c >> 2; kk[j] = (c & 3) * 8; }

#if !ASYNC_LDS
  u32x4 stA[2], stB[2];
#endif

  auto issueTile = [&](int k0, int buf) {
#pragma unroll
    for (int j = 0; j < 2; ++j) {
      const unsigned short* ga = A + (size_t)(rowBase + rr[j]) * DD + k0 + kk[j];
      const unsigned short* gb = B + (size_t)(colBase + rr[j]) * DD + k0 + kk[j];
#if ASYNC_LDS
      g2l_b128(&As[buf][rr[j] * LDSROW + kk[j]], ga);
      g2l_b128(&Bs[buf][rr[j] * LDSROW + kk[j]], gb);
#else
      stA[j] = *(const u32x4*)ga;
      stB[j] = *(const u32x4*)gb;
#endif
    }
  };
  auto commitTile = [&](int buf) {
#if ASYNC_LDS
    async_wait0();
    (void)buf;
#else
#pragma unroll
    for (int j = 0; j < 2; ++j) {
      *(u32x4*)&As[buf][rr[j] * LDSROW + kk[j]] = stA[j];
      *(u32x4*)&Bs[buf][rr[j] * LDSROW + kk[j]] = stB[j];
    }
#endif
  };

  v8f acc[4][2] = {};
  const int rsel = lane & 15, kb = (lane >> 4) * 8;
  auto compute = [&](int buf) {
    v16bf a[4], b[2];
#pragma unroll
    for (int m = 0; m < 4; ++m)
      a[m] = load_frag(&As[buf][(wm * 64 + m * 16 + rsel) * LDSROW], kb);
#pragma unroll
    for (int n = 0; n < 2; ++n)
      b[n] = load_frag(&Bs[buf][(wn * 32 + n * 16 + rsel) * LDSROW], kb);
#pragma unroll
    for (int m = 0; m < 4; ++m)
#pragma unroll
      for (int n = 0; n < 2; ++n)
        acc[m][n] = __builtin_amdgcn_wmma_f32_16x16x32_bf16(
            false, a[m], false, b[n], (short)0, acc[m][n], false, false);
  };

  issueTile(0, 0);
  commitTile(0);
  __syncthreads();
  for (int k0 = 0; k0 < DD; k0 += KSTEP) {
    const int cur = (k0 / KSTEP) & 1;
    const bool more = (k0 + KSTEP) < DD;
    if (more) issueTile(k0 + KSTEP, cur ^ 1);
    if (k0 + 2 * KSTEP < DD) {
      __builtin_prefetch(A + (size_t)(rowBase + (t >> 1)) * DD + k0 + 2 * KSTEP, 0, 1);
      __builtin_prefetch(B + (size_t)(colBase + (t >> 1)) * DD + k0 + 2 * KSTEP, 0, 1);
    }
    compute(cur);
    if (more) commitTile(cur ^ 1);
    __syncthreads();
  }

  // Epilogue: C/D layout -> row = base + v + 8*(lane>>4), col = base + (lane&15)
  const int rhalf = (lane >> 4) * 8;
#pragma unroll
  for (int m = 0; m < 4; ++m)
#pragma unroll
    for (int n = 0; n < 2; ++n) {
      int col = colBase + wn * 32 + n * 16 + rsel;
#pragma unroll
      for (int v = 0; v < 8; ++v) {
        int row = rowBase + wm * 64 + m * 16 + rhalf + v;
        float val = acc[m][n][v];
        size_t idx = (size_t)row * NN + col;
        if (mode == 0) outB[idx] = f32_bf16(val);
        else           outF[idx] = 1.0f - val;
      }
    }
}

// ---------------------------------------------------------------------------
// GEMM variant 2: D = A * B, K = 4096 (sigma1@P, T@sigma2).
// A tile async/direct copy; B tile register-staged + transposed scatter so
// fragment loads match variant 1. mode 0: bf16 out; mode 1: C -= scale*acc.
// ---------------------------------------------------------------------------
__global__ __launch_bounds__(256)
void gemm_ab_k4096(const unsigned short* __restrict__ A,
                   const unsigned short* __restrict__ B,
                   unsigned short* __restrict__ outB,
                   float* __restrict__ outF, int mode, float scale) {
  __shared__ unsigned short As[2][TILE * LDSROW];
  __shared__ unsigned short Bs[2][TILE * LDSROW]; // Bs[col][k] (transposed tile)
  const int t = threadIdx.x, lane = t & 31, wid = t >> 5;
  const int wm = wid >> 2, wn = wid & 3;
  const int rowBase = blockIdx.y * TILE, colBase = blockIdx.x * TILE;

  int rr[2], kk[2];
#pragma unroll
  for (int j = 0; j < 2; ++j) { int c = t + 256 * j; rr[j] = c >> 2; kk[j] = (c & 3) * 8; }
  const int bk = (t >> 4) * 2;  // 0..30 (even) k-row pair in tile
  const int bc = (t & 15) * 8;  // 0..120 col chunk

#if !ASYNC_LDS
  u32x4 stA[2];
#endif
  u32x4 stB0, stB1;

  auto issueTile = [&](int k0, int buf) {
#pragma unroll
    for (int j = 0; j < 2; ++j) {
      const unsigned short* ga = A + (size_t)(rowBase + rr[j]) * NN + k0 + kk[j];
#if ASYNC_LDS
      g2l_b128(&As[buf][rr[j] * LDSROW + kk[j]], ga);
#else
      stA[j] = *(const u32x4*)ga;
#endif
    }
    const unsigned short* gb = B + (size_t)(k0 + bk) * NN + colBase + bc;
    stB0 = *(const u32x4*)gb;
    stB1 = *(const u32x4*)(gb + NN);
  };
  auto commitTile = [&](int buf) {
#if ASYNC_LDS
    async_wait0();
#else
#pragma unroll
    for (int j = 0; j < 2; ++j)
      *(u32x4*)&As[buf][rr[j] * LDSROW + kk[j]] = stA[j];
#endif
#pragma unroll
    for (int jj = 0; jj < 8; ++jj) { // transpose scatter: k,k+1 packed per b32
      unsigned w0 = stB0[jj >> 1], w1 = stB1[jj >> 1];
      unsigned sh = (jj & 1) * 16;
      unsigned pack = ((w0 >> sh) & 0xFFFFu) | (((w1 >> sh) & 0xFFFFu) << 16);
      *(unsigned*)&Bs[buf][(bc + jj) * LDSROW + bk] = pack;
    }
  };

  v8f acc[4][2] = {};
  const int rsel = lane & 15, kb = (lane >> 4) * 8;
  auto compute = [&](int buf) {
    v16bf a[4], b[2];
#pragma unroll
    for (int m = 0; m < 4; ++m)
      a[m] = load_frag(&As[buf][(wm * 64 + m * 16 + rsel) * LDSROW], kb);
#pragma unroll
    for (int n = 0; n < 2; ++n)
      b[n] = load_frag(&Bs[buf][(wn * 32 + n * 16 + rsel) * LDSROW], kb);
#pragma unroll
    for (int m = 0; m < 4; ++m)
#pragma unroll
      for (int n = 0; n < 2; ++n)
        acc[m][n] = __builtin_amdgcn_wmma_f32_16x16x32_bf16(
            false, a[m], false, b[n], (short)0, acc[m][n], false, false);
  };

  issueTile(0, 0);
  commitTile(0);
  __syncthreads();
  for (int k0 = 0; k0 < NN; k0 += KSTEP) {
    const int cur = (k0 / KSTEP) & 1;
    const bool more = (k0 + KSTEP) < NN;
    if (more) issueTile(k0 + KSTEP, cur ^ 1);
    if (k0 + 2 * KSTEP < NN) {
      __builtin_prefetch(A + (size_t)(rowBase + (t >> 1)) * NN + k0 + 2 * KSTEP, 0, 1);
      __builtin_prefetch(B + (size_t)(k0 + 2 * KSTEP + (t >> 4)) * NN + colBase + bc, 0, 1);
    }
    compute(cur);
    if (more) commitTile(cur ^ 1);
    __syncthreads();
  }

  const int rhalf = (lane >> 4) * 8;
#pragma unroll
  for (int m = 0; m < 4; ++m)
#pragma unroll
    for (int n = 0; n < 2; ++n) {
      int col = colBase + wn * 32 + n * 16 + rsel;
#pragma unroll
      for (int v = 0; v < 8; ++v) {
        int row = rowBase + wm * 64 + m * 16 + rhalf + v;
        float val = acc[m][n][v];
        size_t idx = (size_t)row * NN + col;
        if (mode == 0) outB[idx] = f32_bf16(val);
        else           outF[idx] = outF[idx] - scale * val;
      }
    }
}

// ---------------------------------------------------------------------------
// Row-wise softmax: P = softmax(-C/reg) per row (one block per row).
// ---------------------------------------------------------------------------
__global__ __launch_bounds__(256)
void softmax_rows_kernel(const float* __restrict__ C, float* __restrict__ P,
                         float negInvReg) {
  const int row = blockIdx.x;
  const int t = threadIdx.x;
  const float* c = C + (size_t)row * NN;
  float* p = P + (size_t)row * NN;
  __shared__ float red[256];
  float vals[16];
  float mx = -3.4e38f;
#pragma unroll
  for (int i = 0; i < 16; ++i) {
    float l = negInvReg * c[t + i * 256];
    vals[i] = l; mx = fmaxf(mx, l);
  }
  red[t] = mx; __syncthreads();
  for (int s = 128; s > 0; s >>= 1) { if (t < s) red[t] = fmaxf(red[t], red[t + s]); __syncthreads(); }
  mx = red[0]; __syncthreads();
  float sum = 0.0f;
#pragma unroll
  for (int i = 0; i < 16; ++i) { vals[i] = expf(vals[i] - mx); sum += vals[i]; }
  red[t] = sum; __syncthreads();
  for (int s = 128; s > 0; s >>= 1) { if (t < s) red[t] += red[t + s]; __syncthreads(); }
  float inv = 1.0f / red[0];
#pragma unroll
  for (int i = 0; i < 16; ++i) p[t + i * 256] = vals[i] * inv;
}

// ---------------------------------------------------------------------------
// Sinkhorn pieces. Cost C' = 1-C, so P0 = exp(-C') = exp(C-1).
// ---------------------------------------------------------------------------
__global__ __launch_bounds__(256)
void rownorm_exp_kernel(const float* __restrict__ C, float* __restrict__ P) {
  const int row = blockIdx.x;
  const int t = threadIdx.x;
  const float* c = C + (size_t)row * NN;
  float* p = P + (size_t)row * NN;
  __shared__ float red[256];
  float vals[16];
  float sum = 0.0f;
#pragma unroll
  for (int i = 0; i < 16; ++i) { vals[i] = expf(c[t + i * 256] - 1.0f); sum += vals[i]; }
  red[t] = sum; __syncthreads();
  for (int s = 128; s > 0; s >>= 1) { if (t < s) red[t] += red[t + s]; __syncthreads(); }
  float inv = 1.0f / red[0];
#pragma unroll
  for (int i = 0; i < 16; ++i) p[t + i * 256] = vals[i] * inv;
}

__global__ __launch_bounds__(256)
void colsum_kernel(const float* __restrict__ P, float* __restrict__ colsum) {
  const int col = blockIdx.x * 256 + threadIdx.x;
  const int r0 = blockIdx.y * 128;
  float s = 0.0f;
  for (int i = 0; i < 128; ++i) s += P[(size_t)(r0 + i) * NN + col];
  atomicAdd(&colsum[col], s);
}

// fused: col-normalize (iter k) + row-normalize (iter k+1)
__global__ __launch_bounds__(256)
void colrow_norm_kernel(float* __restrict__ P, const float* __restrict__ colsum) {
  const int row = blockIdx.x;
  const int t = threadIdx.x;
  float* p = P + (size_t)row * NN;
  __shared__ float red[256];
  float vals[16];
  float sum = 0.0f;
#pragma unroll
  for (int i = 0; i < 16; ++i) {
    int j = t + i * 256;
    float v = p[j] / colsum[j];
    vals[i] = v; sum += v;
  }
  red[t] = sum; __syncthreads();
  for (int s = 128; s > 0; s >>= 1) { if (t < s) red[t] += red[t + s]; __syncthreads(); }
  float inv = 1.0f / red[0];
#pragma unroll
  for (int i = 0; i < 16; ++i) p[t + i * 256] = vals[i] * inv;
}

__global__ void coldiv_kernel(float* __restrict__ P, const float* __restrict__ colsum) {
  int i = blockIdx.x * blockDim.x + threadIdx.x;
  int stride = gridDim.x * blockDim.x;
  for (; i < NN * NN; i += stride) P[i] /= colsum[i & (NN - 1)];
}

// ---------------------------------------------------------------------------
// loss = -mean_i log_softmax(P)[i, labels[i]]
// ---------------------------------------------------------------------------
__global__ __launch_bounds__(256)
void loss_kernel(const float* __restrict__ P, const int* __restrict__ labels,
                 float* __restrict__ out) {
  const int row = blockIdx.x;
  const int t = threadIdx.x;
  const float* p = P + (size_t)row * NN;
  __shared__ float red[256];
  float mx = -3.4e38f;
#pragma unroll
  for (int i = 0; i < 16; ++i) mx = fmaxf(mx, p[t + i * 256]);
  red[t] = mx; __syncthreads();
  for (int s = 128; s > 0; s >>= 1) { if (t < s) red[t] = fmaxf(red[t], red[t + s]); __syncthreads(); }
  mx = red[0]; __syncthreads();
  float sum = 0.0f;
#pragma unroll
  for (int i = 0; i < 16; ++i) sum += expf(p[t + i * 256] - mx);
  red[t] = sum; __syncthreads();
  for (int s = 128; s > 0; s >>= 1) { if (t < s) red[t] += red[t + s]; __syncthreads(); }
  if (t == 0) {
    int lab = labels[row];
    float logp = p[lab] - mx - logf(red[0]);
    atomicAdd(out, -logp * (1.0f / (float)NN));
  }
}

// ---------------------------------------------------------------------------
// Orchestration
// ---------------------------------------------------------------------------
extern "C" void kernel_launch(void* const* d_in, const int* in_sizes, int n_in,
                              void* d_out, int out_size, void* d_ws, size_t ws_size,
                              hipStream_t stream) {
  const float* img    = (const float*)d_in[0];
  const float* txt    = (const float*)d_in[1];
  const int*   labels = (const int*)d_in[2];
  float* out = (float*)d_out;

  char* ws = (char*)d_ws;
  size_t off = 0;
  auto alloc = [&](size_t bytes) -> void* {
    void* p = ws + off;
    off += (bytes + 255) & ~(size_t)255;
    return p;
  };
  unsigned short* imgB    = (unsigned short*)alloc((size_t)NN * DD * 2);
  unsigned short* txtB    = (unsigned short*)alloc((size_t)NN * DD * 2);
  unsigned short* sigma1B = (unsigned short*)alloc((size_t)NN * NN * 2);
  unsigned short* sigma2B = (unsigned short*)alloc((size_t)NN * NN * 2);
  unsigned short* Pb      = (unsigned short*)alloc((size_t)NN * NN * 2);
  unsigned short* Tb      = (unsigned short*)alloc((size_t)NN * NN * 2);
  float* Cmat   = (float*)alloc((size_t)NN * NN * 4);
  float* P      = (float*)alloc((size_t)NN * NN * 4);
  float* colsum = (float*)alloc((size_t)NN * 4);

  const dim3 blk(256);
  const dim3 g2d(NN / TILE, NN / TILE); // 32x32

  // bf16 copies of features
  cvt_bf16_kernel<<<1024, blk, 0, stream>>>(img, imgB, NN * DD);
  cvt_bf16_kernel<<<1024, blk, 0, stream>>>(txt, txtB, NN * DD);

  // sigma1 = img@img^T, sigma2 = txt@txt^T (bf16), C = 1 - img@txt^T (f32)
  gemm_abt_k512<<<g2d, blk, 0, stream>>>(imgB, imgB, sigma1B, nullptr, 0);
  gemm_abt_k512<<<g2d, blk, 0, stream>>>(txtB, txtB, sigma2B, nullptr, 0);
  gemm_abt_k512<<<g2d, blk, 0, stream>>>(imgB, txtB, nullptr, Cmat, 1);

  // P = softmax(-C/reg), reg = 0.01
  softmax_rows_kernel<<<NN, blk, 0, stream>>>(Cmat, P, -100.0f);

  for (int it = 0; it < 5; ++it) {
    // C -= (a1*a2) * sigma1 @ P @ sigma2
    cvt_bf16_kernel<<<2048, blk, 0, stream>>>(P, Pb, NN * NN);
    gemm_ab_k4096<<<g2d, blk, 0, stream>>>(sigma1B, Pb, Tb, nullptr, 0, 0.0f);
    gemm_ab_k4096<<<g2d, blk, 0, stream>>>(Tb, sigma2B, nullptr, Cmat, 1, 1.0e-4f);

    // P = sinkhorn(1 - C, 5 iters): exp+row, then 5x (colsum; col[+row])
    rownorm_exp_kernel<<<NN, blk, 0, stream>>>(Cmat, P);
    for (int s = 0; s < 5; ++s) {
      zero_kernel<<<16, blk, 0, stream>>>(colsum, NN);
      colsum_kernel<<<dim3(16, 32), blk, 0, stream>>>(P, colsum);
      if (s < 4) colrow_norm_kernel<<<NN, blk, 0, stream>>>(P, colsum);
      else       coldiv_kernel<<<4096, blk, 0, stream>>>(P, colsum);
    }
  }

  // loss = -mean log_softmax(P)[i, labels[i]]
  zero_kernel<<<1, 32, 0, stream>>>(out, 1);
  loss_kernel<<<NN, blk, 0, stream>>>(P, labels, out);
}
